// AttentionSequencePoolingLayer_30296699306118
// MI455X (gfx1250) — matrix operand order repaired and checked by reading, not compiled, not run on hardware
//
#include <hip/hip_runtime.h>

typedef __attribute__((ext_vector_type(16))) _Float16 v16h;
typedef __attribute__((ext_vector_type(8)))  _Float16 v8h;
typedef __attribute__((ext_vector_type(4)))  _Float16 v4h;
typedef __attribute__((ext_vector_type(2)))  _Float16 v2h;
typedef __attribute__((ext_vector_type(8)))  float    v8f;

#define B_TOTAL   4096
#define T_SEQ     200
#define T_PAD     208          // 13 WMMA tiles of 16
#define C_DIM     64
#define STRIDE    72           // LDS row stride in halves (144B) -> bank-conflict-free A loads
#define BPB       2            // batches per block (1 wave32 per batch)

__global__ __launch_bounds__(32 * BPB)
void attn_pool_kernel(const float* __restrict__ q,
                      const float* __restrict__ keys,
                      const int*   __restrict__ klen,
                      const float* __restrict__ W,
                      const float* __restrict__ bias,
                      float*       __restrict__ out)
{
    __shared__ _Float16 keysh[BPB][T_PAD * STRIDE];
    __shared__ float    scores[BPB][T_PAD];
    __shared__ _Float16 Wh[C_DIM];

    const int tid  = threadIdx.x;
    const int wave = tid >> 5;
    const int lane = tid & 31;
    const int b    = blockIdx.x * BPB + wave;

    // ---- stage W_k (keys half of W) as f16 into LDS (both waves write identical values)
    Wh[lane]      = (_Float16)W[C_DIM + lane];
    Wh[lane + 32] = (_Float16)W[C_DIM + 32 + lane];

    // ---- stage keys[b] into LDS as f16, coalesced 512B/wave per iteration
    const float* kb = keys + (size_t)b * (T_SEQ * C_DIM);
    for (int i = 0; i < (T_SEQ * C_DIM) / 128; ++i) {
        const int flat = i * 128 + lane * 4;
        const float4 v = *(const float4*)(kb + flat);
        const int r = 2 * i + (lane >> 4);
        const int c = (lane & 15) * 4;
        v4h h = { (_Float16)v.x, (_Float16)v.y, (_Float16)v.z, (_Float16)v.w };
        *(v4h*)&keysh[wave][r * STRIDE + c] = h;
    }
    // zero-pad rows 200..207 (cols 0..63), 16 halves per lane
    {
        const int r = T_SEQ + (lane >> 2);
        const int c = (lane & 3) * 16;
        v8h z = {};
        *(v8h*)&keysh[wave][r * STRIDE + c]     = z;
        *(v8h*)&keysh[wave][r * STRIDE + c + 8] = z;
    }
    __syncthreads();

    // ---- raw scores via WMMA: D(16x16) = keys_tile(16x64) x Wk(64x16 replicated cols)
    // A layout (16-bit 16x32): lanes 0-15 hold K {0..7,16..23}, lanes 16-31 hold K {8..15,24..31}
    const int offA = (lane < 16) ? 0 : 8;
    const int rowA = (lane & 15);
    // B layout (16-bit 32x16): lanes 0-15 hold K {0..7,16..23}, lanes 16-31 hold K {8..15,24..31}
    const int kbB  = (lane < 16) ? 0 : 8;

    for (int tb = 0; tb < T_PAD / 16; ++tb) {
        v8f acc = {0.f, 0.f, 0.f, 0.f, 0.f, 0.f, 0.f, 0.f};
#pragma unroll
        for (int k0 = 0; k0 < C_DIM; k0 += 32) {
            const _Float16* pa = &keysh[wave][(tb * 16 + rowA) * STRIDE + k0 + offA];
            v8h alo = *(const v8h*)pa;
            v8h ahi = *(const v8h*)(pa + 16);
            v16h a = __builtin_shufflevector(alo, ahi,
                     0,1,2,3,4,5,6,7,8,9,10,11,12,13,14,15);
            const _Float16* pb = &Wh[k0 + kbB];
            v8h blo = *(const v8h*)pb;
            v8h bhi = *(const v8h*)(pb + 16);
            v16h bf = __builtin_shufflevector(blo, bhi,
                     0,1,2,3,4,5,6,7,8,9,10,11,12,13,14,15);
            acc = __builtin_amdgcn_wmma_f32_16x16x32_f16(
                      false, a, false, bf, (short)0, acc, false, false);
        }
        // D: lane n, VGPR r -> M = r + (lane<16?0:8), N = n&15.  Keep column N==0.
        if ((lane & 15) == 0) {
            const int mb = tb * 16 + (lane >> 4) * 8;
#pragma unroll
            for (int r = 0; r < 8; ++r) scores[wave][mb + r] = acc[r];
        }
    }
    __syncthreads();

    // ---- qdot = dot(q_b, W_q) + bias (wave reduction)
    float part = q[(size_t)b * C_DIM + lane]      * W[lane]
               + q[(size_t)b * C_DIM + 32 + lane] * W[32 + lane];
#pragma unroll
    for (int m = 16; m >= 1; m >>= 1) part += __shfl_xor(part, m, 32);
    const float qdot = part + bias[0];
    const int   len  = klen[b];

    // ---- tanh + mask (exact reference pad value), wave max
    const float MASKP = -4294967295.0f;   // -(2^32)+1
    float lmax = -3.0e38f;
#pragma unroll
    for (int i = 0; i < 7; ++i) {
        const int t = lane + 32 * i;
        if (t < T_PAD) {
            float s;
            if (t < len)        s = tanhf(scores[wave][t] + qdot);
            else if (t < T_SEQ) s = MASKP;
            else                s = -3.0e38f;   // pad rows: exp -> 0 always
            scores[wave][t] = s;
            lmax = fmaxf(lmax, s);
        }
    }
#pragma unroll
    for (int m = 16; m >= 1; m >>= 1) lmax = fmaxf(lmax, __shfl_xor(lmax, m, 32));

    // ---- exp + wave sum (leave unnormalized weights in LDS)
    float lsum = 0.f;
#pragma unroll
    for (int i = 0; i < 7; ++i) {
        const int t = lane + 32 * i;
        if (t < T_PAD) {
            const float e = __expf(scores[wave][t] - lmax);
            scores[wave][t] = e;
            lsum += e;
        }
    }
#pragma unroll
    for (int m = 16; m >= 1; m >>= 1) lsum += __shfl_xor(lsum, m, 32);
    const float inv = 1.0f / lsum;
    __syncthreads();

    // ---- out[b,c] = (1/sum) * sum_t e[t] * keys[t,c]; each lane owns channels 2*lane, 2*lane+1
    float a0 = 0.f, a1 = 0.f;
    const _Float16* kh = &keysh[wave][0];
#pragma unroll 4
    for (int t = 0; t < T_SEQ; ++t) {
        const float w = scores[wave][t];
        const v2h kv = *(const v2h*)&kh[t * STRIDE + 2 * lane];
        a0 += w * (float)kv[0];
        a1 += w * (float)kv[1];
    }
    out[(size_t)b * C_DIM + 2 * lane]     = a0 * inv;
    out[(size_t)b * C_DIM + 2 * lane + 1] = a1 * inv;
}

extern "C" void kernel_launch(void* const* d_in, const int* in_sizes, int n_in,
                              void* d_out, int out_size, void* d_ws, size_t ws_size,
                              hipStream_t stream) {
    (void)in_sizes; (void)n_in; (void)out_size; (void)d_ws; (void)ws_size;
    const float* q    = (const float*)d_in[0];
    const float* keys = (const float*)d_in[1];
    const int*   kl   = (const int*)d_in[2];
    const float* W    = (const float*)d_in[3];
    const float* bias = (const float*)d_in[4];
    float* out = (float*)d_out;

    dim3 grid(B_TOTAL / BPB);
    dim3 block(32 * BPB);
    hipLaunchKernelGGL(attn_pool_kernel, grid, block, 0, stream,
                       q, keys, kl, W, bias, out);
}
